// SelfAttention_4750233829377
// MI455X (gfx1250) — matrix operand order repaired and checked by reading, not compiled, not run on hardware
//
#include <hip/hip_runtime.h>
#include <hip/hip_bf16.h>

typedef __attribute__((ext_vector_type(16))) _Float16 v16h;
typedef __attribute__((ext_vector_type(8)))  _Float16 v8h;
typedef __attribute__((ext_vector_type(8)))  float    v8f;

#define B_  4
#define S_  2048
#define D_  1024
#define H_  16
#define HD_ 64
#define NH_ (H_ * HD_)   // 1024

// D = A(16x32 f16) * B(32x16 f16) + C(16x16 f32), wave32.
__device__ __forceinline__ v8f wmma_f16(v16h a, v16h b, v8f c) {
  return __builtin_amdgcn_wmma_f32_16x16x32_f16(
      /*neg_a=*/false, a, /*neg_b=*/false, b,
      /*c_mod=*/(short)0, c, /*reuse_a=*/false, /*reuse_b=*/false);
}

// A-fragment (16x32, 16-bit), contiguous k order (B-frags use the same):
//   lane<16 : row M=lane,    halves 0..15 = K k0..k0+15
//   lane>=16: row M=lane-16, halves 0..15 = K k0+16..k0+31
__device__ __forceinline__ v16h load_a_frag(const _Float16* row, int k0, int half) {
  const _Float16* p = row + k0 + 16 * half;
  v8h lo = *(const v8h*)(p);
  v8h hi = *(const v8h*)(p + 8);
  v16h r;
#pragma unroll
  for (int i = 0; i < 8; ++i) { r[i] = lo[i]; r[i + 8] = hi[i]; }
  return r;
}

// ---------------------------------------------------------------------------
// fp32 -> fp16 elementwise convert
__global__ __launch_bounds__(256)
void cvt_f32_to_f16(const float* __restrict__ src, _Float16* __restrict__ dst, int n) {
  int i = blockIdx.x * 256 + threadIdx.x;
  if (i < n) dst[i] = (_Float16)src[i];
}

// W[K][N] fp32 -> WT[N][K] fp16  (so WMMA B-fragments are contiguous row reads)
__global__ __launch_bounds__(256)
void transpose_cvt(const float* __restrict__ W, _Float16* __restrict__ WT, int K, int N) {
  int i = blockIdx.x * 256 + threadIdx.x;
  if (i < K * N) {
    int k = i / N;
    int n = i - k * N;
    WT[(size_t)n * K + k] = (_Float16)W[i];
  }
}

// ---------------------------------------------------------------------------
// C[M,N] = A[M,K] * BT[N,K]^T   (A row-major f16, B given transposed f16)
// Block: 256 thr = 8 waves arranged 4x2; wave tile 32x64 (2x4 WMMA frags,
// 8 WMMA per K=32 step). Block tile 128x128.
// mode 0: fp32 C[M,N]
// mode 1: fused-QKV scatter. N==3072; n>>10 selects Q/K/V. Cout points at the
//         QKV workspace base: Q f16 [B,H,S,HD] at +0, K at +8M halfs,
//         V^T f16 [B,H,HD,S] at +16M halfs.
__global__ __launch_bounds__(256)
void gemm_nt(const _Float16* __restrict__ A, const _Float16* __restrict__ BT,
             void* __restrict__ Cout, int M, int N, int K, int mode) {
  const int lane = threadIdx.x & 31;
  const int wave = threadIdx.x >> 5;
  const int half = lane >> 4;
  const int l16  = lane & 15;
  const int bm = blockIdx.x * 128 + (wave >> 1) * 32;
  const int bn = blockIdx.y * 128 + (wave & 1) * 64;

  const _Float16* a0 = A + (size_t)(bm      + l16) * K;
  const _Float16* a1 = A + (size_t)(bm + 16 + l16) * K;
  const _Float16* b0 = BT + (size_t)(bn      + l16) * K;
  const _Float16* b1 = BT + (size_t)(bn + 16 + l16) * K;
  const _Float16* b2 = BT + (size_t)(bn + 32 + l16) * K;
  const _Float16* b3 = BT + (size_t)(bn + 48 + l16) * K;

  v8f acc[2][4] = {};
  for (int k0 = 0; k0 < K; k0 += 32) {
    // prefetch ~8 iterations ahead (speculative; drops silently past end)
    __builtin_prefetch(a0 + k0 + 256, 0, 1);
    __builtin_prefetch(b0 + k0 + 256, 0, 1);
    __builtin_prefetch(b2 + k0 + 256, 0, 1);

    v16h fa0 = load_a_frag(a0, k0, half);
    v16h fa1 = load_a_frag(a1, k0, half);
    // B-fragment (32x16): lane<16 -> col=l16, K k0..k0+15 ; lane>=16 -> K k0+16..31
    v16h fb0 = *(const v16h*)(b0 + k0 + 16 * half);
    v16h fb1 = *(const v16h*)(b1 + k0 + 16 * half);
    v16h fb2 = *(const v16h*)(b2 + k0 + 16 * half);
    v16h fb3 = *(const v16h*)(b3 + k0 + 16 * half);
    acc[0][0] = wmma_f16(fa0, fb0, acc[0][0]);
    acc[0][1] = wmma_f16(fa0, fb1, acc[0][1]);
    acc[0][2] = wmma_f16(fa0, fb2, acc[0][2]);
    acc[0][3] = wmma_f16(fa0, fb3, acc[0][3]);
    acc[1][0] = wmma_f16(fa1, fb0, acc[1][0]);
    acc[1][1] = wmma_f16(fa1, fb1, acc[1][1]);
    acc[1][2] = wmma_f16(fa1, fb2, acc[1][2]);
    acc[1][3] = wmma_f16(fa1, fb3, acc[1][3]);
  }

  if (mode == 0) {
    float* C = (float*)Cout;
#pragma unroll
    for (int mi = 0; mi < 2; ++mi)
#pragma unroll
      for (int ni = 0; ni < 4; ++ni) {
        const int nn = bn + 16 * ni + l16;
#pragma unroll
        for (int i = 0; i < 8; ++i) {
          const int mm = bm + 16 * mi + i + 8 * half;  // C-frag row map
          C[(size_t)mm * N + nn] = acc[mi][ni][i];
        }
      }
  } else {
    _Float16* C = (_Float16*)Cout;
    const size_t QKV_STRIDE = (size_t)B_ * H_ * S_ * HD_;  // 8M halfs per tensor
#pragma unroll
    for (int mi = 0; mi < 2; ++mi)
#pragma unroll
      for (int ni = 0; ni < 4; ++ni) {
        const int nn    = bn + 16 * ni + l16;
        const int which = nn >> 10;          // 0=Q, 1=K, 2=V
        const int rem   = nn & (NH_ - 1);
        const int hh    = rem >> 6;
        const int dd    = rem & (HD_ - 1);
#pragma unroll
        for (int i = 0; i < 8; ++i) {
          const int mm = bm + 16 * mi + i + 8 * half;
          const int bb = mm >> 11;           // mm / S_
          const int ss = mm & (S_ - 1);
          size_t idx;
          if (which < 2)                      // Q / K : [B,H,S,HD]
            idx = (size_t)which * QKV_STRIDE +
                  (((size_t)(bb * H_ + hh) * S_) + ss) * HD_ + dd;
          else                                // V^T : [B,H,HD,S]
            idx = 2 * QKV_STRIDE +
                  (((size_t)(bb * H_ + hh) * HD_) + dd) * S_ + ss;
          C[idx] = (_Float16)acc[mi][ni][i];
        }
      }
  }
}

// ---------------------------------------------------------------------------
// Causal flash attention. Grid: (B*H, S/128), 256 threads = 8 waves.
// Each wave owns 16 query rows; streams keys in tiles of 32 with online softmax
// in the pre-scaled base-2 domain (scores *= 1/sqrt(HD)*log2e up front).
// Qh/Kh: [B,H,S,HD] f16 ; VTh: [B,H,HD,S] f16 ; Oh: [B,S,H*HD] f16.
__global__ __launch_bounds__(256)
void attn_fwd(const _Float16* __restrict__ Qh, const _Float16* __restrict__ Kh,
              const _Float16* __restrict__ VTh, _Float16* __restrict__ Oh) {
  __shared__ _Float16 ldsP[8][16][32];   // per-wave P tile (C-layout -> A-layout)

  const int lane = threadIdx.x & 31;
  const int wave = threadIdx.x >> 5;
  const int half = lane >> 4;
  const int l16  = lane & 15;
  const int b = blockIdx.x / H_;
  const int h = blockIdx.x % H_;
  const int q0 = blockIdx.y * 128 + wave * 16;

  const _Float16* Q  = Qh  + (size_t)(b * H_ + h) * S_ * HD_;
  const _Float16* Kp = Kh  + (size_t)(b * H_ + h) * S_ * HD_;
  const _Float16* VT = VTh + (size_t)(b * H_ + h) * HD_ * S_;

  const _Float16* qrow = Q + (size_t)(q0 + l16) * HD_;
  const v16h qa0 = load_a_frag(qrow, 0,  half);   // d 0..31
  const v16h qa1 = load_a_frag(qrow, 32, half);   // d 32..63

  v8f o0 = {}, o1 = {}, o2 = {}, o3 = {};
  float mrow[8], lrow[8];
#pragma unroll
  for (int i = 0; i < 8; ++i) { mrow[i] = -1e30f; lrow[i] = 0.0f; }

  const float SCL2E = 0.125f * 1.44269504f;  // 1/sqrt(HD) * log2(e)

  const int kend = q0 + 16;  // causal: keys beyond q0+15 never contribute
  for (int kb = 0; kb < kend; kb += 32) {
    const _Float16* krow0 = Kp + (size_t)(kb      + l16) * HD_;
    const _Float16* krow1 = Kp + (size_t)(kb + 16 + l16) * HD_;
    // prefetch next key tile + next V^T columns while this tile computes
    __builtin_prefetch(krow0 + 32 * HD_, 0, 1);
    __builtin_prefetch(krow1 + 32 * HD_, 0, 1);
    __builtin_prefetch(VT + (size_t)l16 * S_ + kb + 32, 0, 1);

    // ---- scores: Q(16x64) x K^T(64x32) as 4 WMMA into two 16x16 frags
    const v16h kb00 = *(const v16h*)(krow0 + 16 * half);        // keys kb..+15, d 0..31
    const v16h kb01 = *(const v16h*)(krow0 + 32 + 16 * half);   // keys kb..+15, d 32..63
    const v16h kb10 = *(const v16h*)(krow1 + 16 * half);        // keys kb+16..+31
    const v16h kb11 = *(const v16h*)(krow1 + 32 + 16 * half);
    v8f s0 = {}, s1 = {};
    s0 = wmma_f16(qa0, kb00, s0);
    s0 = wmma_f16(qa1, kb01, s0);
    s1 = wmma_f16(qa0, kb10, s1);
    s1 = wmma_f16(qa1, kb11, s1);

    const bool diag = (kb + 31 > q0);
#pragma unroll
    for (int i = 0; i < 8; ++i) {
      float v0 = s0[i] * SCL2E;     // base-2 domain
      float v1 = s1[i] * SCL2E;
      if (diag) {
        const int qi = q0 + i + 8 * half;
        if (kb      + l16 > qi) v0 = -1e30f;
        if (kb + 16 + l16 > qi) v1 = -1e30f;
      }
      float rm = fmaxf(v0, v1);
#pragma unroll
      for (int off = 1; off < 16; off <<= 1)
        rm = fmaxf(rm, __shfl_xor(rm, off, 16));
      const float mnew  = fmaxf(mrow[i], rm);
      const float p0    = exp2f(v0 - mnew);
      const float p1    = exp2f(v1 - mnew);
      const float alpha = exp2f(mrow[i] - mnew);
      float rs = p0 + p1;
#pragma unroll
      for (int off = 1; off < 16; off <<= 1)
        rs += __shfl_xor(rs, off, 16);
      lrow[i] = lrow[i] * alpha + rs;
      mrow[i] = mnew;
      o0[i] *= alpha; o1[i] *= alpha; o2[i] *= alpha; o3[i] *= alpha;
      // C-frag element (row i+8*half, col l16 / l16+16) -> LDS row-major P tile
      ldsP[wave][i + 8 * half][l16]      = (_Float16)p0;
      ldsP[wave][i + 8 * half][l16 + 16] = (_Float16)p1;
    }
    // wave-private LDS tile; DS ops are in-order per wave — just stop reordering
    asm volatile("" ::: "memory");

    // re-read P in A-fragment layout (16x32 over the 32 keys of this tile)
    const _Float16* pp = &ldsP[wave][l16][16 * half];
    const v8h plo = *(const v8h*)(pp);
    const v8h phi = *(const v8h*)(pp + 8);
    v16h pa;
#pragma unroll
    for (int i = 0; i < 8; ++i) { pa[i] = plo[i]; pa[i + 8] = phi[i]; }
    asm volatile("" ::: "memory");

    // ---- O += P(16x32) x V(32x64): B-frags from V^T rows (contiguous in s')
    o0 = wmma_f16(pa, *(const v16h*)(VT + (size_t)(l16     ) * S_ + kb + 16 * half), o0);
    o1 = wmma_f16(pa, *(const v16h*)(VT + (size_t)(l16 + 16) * S_ + kb + 16 * half), o1);
    o2 = wmma_f16(pa, *(const v16h*)(VT + (size_t)(l16 + 32) * S_ + kb + 16 * half), o2);
    o3 = wmma_f16(pa, *(const v16h*)(VT + (size_t)(l16 + 48) * S_ + kb + 16 * half), o3);
  }

  // epilogue: normalize by l and store Oh[b, q, h*64 + d]
#pragma unroll
  for (int i = 0; i < 8; ++i) {
    const int q = q0 + i + 8 * half;
    const float inv = 1.0f / lrow[i];
    const size_t base = ((size_t)b * S_ + q) * NH_ + (size_t)h * HD_;
    Oh[base + l16     ] = (_Float16)(o0[i] * inv);
    Oh[base + l16 + 16] = (_Float16)(o1[i] * inv);
    Oh[base + l16 + 32] = (_Float16)(o2[i] * inv);
    Oh[base + l16 + 48] = (_Float16)(o3[i] * inv);
  }
}

// ---------------------------------------------------------------------------
extern "C" void kernel_launch(void* const* d_in, const int* in_sizes, int n_in,
                              void* d_out, int out_size, void* d_ws, size_t ws_size,
                              hipStream_t stream) {
  (void)in_sizes; (void)n_in; (void)out_size; (void)ws_size;

  const float* x  = (const float*)d_in[0];
  const float* Wq = (const float*)d_in[1];
  const float* Wk = (const float*)d_in[2];
  const float* Wv = (const float*)d_in[3];
  const float* Wo = (const float*)d_in[4];

  char* ws = (char*)d_ws;
  const size_t MB = 1024u * 1024u;
  _Float16* xh    = (_Float16*)(ws +  0 * MB);  // [B*S, D]              16 MB
  _Float16* WqkvT = (_Float16*)(ws + 16 * MB);  // [3072, 1024] f16       6 MB
  _Float16* WoT   = (_Float16*)(ws + 22 * MB);  // [1024, 1024] f16       2 MB
  _Float16* Qh    = (_Float16*)(ws + 24 * MB);  // [B,H,S,HD]            16 MB
  _Float16* Kh    = (_Float16*)(ws + 40 * MB);  // [B,H,S,HD]            16 MB
  _Float16* VTh   = (_Float16*)(ws + 56 * MB);  // [B,H,HD,S]            16 MB
  _Float16* Oh    = (_Float16*)(ws + 72 * MB);  // [B,S,H*HD]            16 MB

  const int NX = B_ * S_ * D_;                  // 8388608
  cvt_f32_to_f16<<<NX / 256, 256, 0, stream>>>(x, xh, NX);
  // fused transposed weight stack: rows [0,1024)=Wq^T, [1024,2048)=Wk^T, [2048,3072)=Wv^T
  transpose_cvt<<<(D_ * NH_) / 256, 256, 0, stream>>>(Wq, WqkvT,                 D_, NH_);
  transpose_cvt<<<(D_ * NH_) / 256, 256, 0, stream>>>(Wk, WqkvT + (size_t)D_ * NH_,     D_, NH_);
  transpose_cvt<<<(D_ * NH_) / 256, 256, 0, stream>>>(Wv, WqkvT + (size_t)2 * D_ * NH_, D_, NH_);
  transpose_cvt<<<(NH_ * NH_) / 256, 256, 0, stream>>>(Wo, WoT, NH_, NH_);

  // single fused QKV projection: [8192,1024] x [1024,3072] -> Q,K,V^T scatter
  dim3 gqkv(B_ * S_ / 128, (3 * NH_) / 128);    // (64, 24)
  gemm_nt<<<gqkv, 256, 0, stream>>>(xh, WqkvT, Qh, B_ * S_, 3 * NH_, D_, 1);

  dim3 gattn(B_ * H_, S_ / 128);                // (64, 16)
  attn_fwd<<<gattn, 256, 0, stream>>>(Qh, Kh, VTh, Oh);

  dim3 gout(B_ * S_ / 128, NH_ / 128);          // (64, 8)
  gemm_nt<<<gout, 256, 0, stream>>>(Oh, WoT, d_out, B_ * S_, NH_, NH_, 0);
}
